// MixedIGAB_3444563771568
// MI455X (gfx1250) — matrix operand order repaired and checked by reading, not compiled, not run on hardware
//
#include <hip/hip_runtime.h>
#include <math.h>

// ---------------------------------------------------------------------------
// MixedIGAB for MI455X (gfx1250): f16 WMMA GEMMs with TDM (tensor_load_to_lds)
// double-buffered weight staging + fused epilogues.
// B=2, C=256, H=W=128, N=16384, BN=32768, HEADS=8, d=32, L=2, MC=1024.
// ---------------------------------------------------------------------------

typedef _Float16 h8  __attribute__((ext_vector_type(8)));
typedef _Float16 h16 __attribute__((ext_vector_type(16)));
typedef float    f8  __attribute__((ext_vector_type(8)));
typedef unsigned int u32x4 __attribute__((ext_vector_type(4)));
typedef int      i32x4v __attribute__((ext_vector_type(4)));
typedef int      i32x8v __attribute__((ext_vector_type(8)));

#define C_    256
#define H_    128
#define W_    128
#define N_    16384
#define B_    2
#define BN_   32768
#define HEADS_ 8
#define MC_   1024
#define L_    2

__device__ __forceinline__ float gelu_f(float x) {
  return 0.5f * x * (1.0f + erff(x * 0.70710678118654752f));
}

// --------- TDM: DMA a 64-row x 128-half weight panel chunk into LDS --------
// D# per CDNA5 ISA. 2D tile, data_size=2B,
// pad_enable: +4 DWORDs after every 64 DWORDs (256B row) -> LDS row stride
// 136 halves (68 DW, == 4 mod 64 banks: conflict-free fragment reads).
__device__ __forceinline__ void tdm_load_panel(const _Float16* g, unsigned lds,
                                               int K) {
  unsigned long long ga = (unsigned long long)(uintptr_t)g;
  u32x4 g0;
  g0[0] = 1u;                                  // count=1 (user descriptor)
  g0[1] = lds;                                 // lds_addr (bytes)
  g0[2] = (unsigned)ga;                        // global_addr[31:0]
  g0[3] = ((unsigned)(ga >> 32) & 0x01ffffffu) // global_addr[56:32]
          | (2u << 30);                        // type=2 ("image")
  unsigned td1 = 1u << 20;                     // generous row bound (in-range)
  i32x8v g1;
  g1[0] = (int)((1u << 16)     // data_size = 2 bytes
                | (1u << 20)   // pad_enable
                | (5u << 22)   // pad_interval = 64 DWORDs (256B)
                | (3u << 25)); // pad_amount   = 4 DWORDs (16B)
  g1[1] = (int)(((unsigned)K & 0xffffu) << 16);                 // tensor_dim0 lo
  g1[2] = (int)((((unsigned)K >> 16) & 0xffffu) |               // tensor_dim0 hi
                ((td1 & 0xffffu) << 16));                       // tensor_dim1 lo
  g1[3] = (int)(((td1 >> 16) & 0xffffu) | (128u << 16));        // tile_dim0=128
  g1[4] = 64;                                                   // tile_dim1=64
  g1[5] = K;                                                    // dim0 stride
  g1[6] = 0;
  g1[7] = 0;
  i32x4v z4 = {0, 0, 0, 0};
#if __clang_major__ >= 23
  i32x8v z8 = {0, 0, 0, 0, 0, 0, 0, 0};
  __builtin_amdgcn_tensor_load_to_lds(g0, g1, z4, z4, z8, 0);
#else
  __builtin_amdgcn_tensor_load_to_lds(g0, g1, z4, z4, 0);
#endif
}

// ---------------- generic tiled 2D transpose (per-batch, 32x32 tiles) ------
template <typename TI, typename TO>
__global__ void k_tr2d(const TI* __restrict__ in, TO* __restrict__ out,
                       int rows, int cols) {
  __shared__ float tile[32][33];
  int bz = blockIdx.z;
  const TI* ip = in + (size_t)bz * rows * cols;
  TO* op = out + (size_t)bz * rows * cols;
  int c0 = blockIdx.x * 32, r0 = blockIdx.y * 32;
#pragma unroll
  for (int j = 0; j < 4; j++) {
    int r = r0 + threadIdx.y + 8 * j;
    tile[threadIdx.y + 8 * j][threadIdx.x] =
        (float)ip[(size_t)r * cols + c0 + threadIdx.x];
  }
  __syncthreads();
#pragma unroll
  for (int j = 0; j < 4; j++) {
    int c = c0 + threadIdx.y + 8 * j;
    op[(size_t)c * rows + r0 + threadIdx.x] =
        (TO)tile[threadIdx.x][threadIdx.y + 8 * j];
  }
}

// ---------------- elementwise f32 -> f16 cast ------------------------------
__global__ void k_cast(const float* __restrict__ in, _Float16* __restrict__ out,
                       int n) {
  int i = blockIdx.x * blockDim.x + threadIdx.x;
  if (i < n) out[i] = (_Float16)in[i];
}

// ---------------- WMMA GEMM: D = A[BNxK] * Bt[NoutxK]^T --------------------
// B panel staged in LDS by TDM, double buffered per 128-half K-chunk.
// Per k-step: all 4 B fragments are preloaded (one ds clause) before the
// 4 WMMAs issue, so LDS latency overlaps matrix-op execution.
// EPI 0: store f16          EPI 1: *aux (illu) then store f16
// EPI 2: GELU then f16      EPI 3: outF += D + bias (residual accumulate)
template <int EPI>
__global__ __launch_bounds__(256)
void k_gemm_wmma(const _Float16* __restrict__ A, const _Float16* __restrict__ Bt,
                 const _Float16* __restrict__ aux, const float* __restrict__ bias,
                 _Float16* __restrict__ out16, float* __restrict__ outF,
                 int K, int Nout) {
  constexpr int KC = 128;           // K-chunk in halves
  constexpr int LROW = KC + 8;      // LDS row stride in halves (TDM padded)
  __shared__ __align__(16) _Float16 sB[2][64 * LROW];
  int lane = threadIdx.x & 31;
  int wave = threadIdx.x >> 5;
  int mBase = (blockIdx.y * 8 + wave) * 16;
  int nBase = blockIdx.x * 64;
  int sub = lane >> 4;  // half-wave select per ISA 16-bit A/B layouts
  const _Float16* Ap = A + (size_t)(mBase + (lane & 15)) * K + sub * 8;
  const _Float16* Bpanel = Bt + (size_t)nBase * K;
  f8 acc[4] = {};
  int nChunks = K / KC;
  if (wave == 0)
    tdm_load_panel(Bpanel, (unsigned)(uintptr_t)(void*)&sB[0][0], K);
  for (int c = 0; c < nChunks; ++c) {
    if (wave == 0) __builtin_amdgcn_s_wait_tensorcnt(0);
    __syncthreads();  // chunk c ready; all waves done reading buf[(c+1)&1]
    if (wave == 0 && c + 1 < nChunks)
      tdm_load_panel(Bpanel + (c + 1) * KC,
                     (unsigned)(uintptr_t)(void*)&sB[(c + 1) & 1][0], K);
    const _Float16* base = &sB[c & 1][0];
    int kG = c * KC;
    // Preload all A fragments of this chunk (one clause, one wait)
    h16 a[4];
#pragma unroll
    for (int j = 0; j < 4; ++j) {
      h8 alo = *(const h8*)(Ap + kG + j * 32);
      h8 ahi = *(const h8*)(Ap + kG + j * 32 + 16);
      a[j] = __builtin_shufflevector(alo, ahi, 0, 1, 2, 3, 4, 5, 6, 7, 8, 9,
                                     10, 11, 12, 13, 14, 15);
    }
    __builtin_prefetch(Ap + kG + KC, 0, 3);  // near-cache prefetch of A stream
#pragma unroll
    for (int j = 0; j < 4; ++j) {
      // Gather the 4 B fragments of this k-step first (ds clause in flight),
      // then issue the 4 WMMAs: waits become overlapping instead of 0.
      h16 b[4];
#pragma unroll
      for (int t = 0; t < 4; ++t) {
        const _Float16* Bp =
            base + (t * 16 + (lane & 15)) * LROW + j * 32 + sub * 8;
        h8 blo = *(const h8*)(Bp);
        h8 bhi = *(const h8*)(Bp + 16);
        b[t] = __builtin_shufflevector(blo, bhi, 0, 1, 2, 3, 4, 5, 6, 7, 8, 9,
                                       10, 11, 12, 13, 14, 15);
      }
#pragma unroll
      for (int t = 0; t < 4; ++t) {
        acc[t] = __builtin_amdgcn_wmma_f32_16x16x32_f16(
            false, a[j], false, b[t], (short)0, acc[t], false, false);
      }
    }
  }
  // C/D layout: VGPR r -> (M = r + 8*sub, N = lane&15)
  int n = lane & 15, mOff = sub * 8;
#pragma unroll
  for (int t = 0; t < 4; t++) {
#pragma unroll
    for (int r = 0; r < 8; r++) {
      float v = acc[t][r];
      size_t row = (size_t)(mBase + mOff + r);
      size_t col = (size_t)(nBase + t * 16 + n);
      size_t idx = row * (size_t)Nout + col;
      if (EPI == 0) {
        out16[idx] = (_Float16)v;
      } else if (EPI == 1) {
        out16[idx] = (_Float16)(v * (float)aux[idx]);
      } else if (EPI == 2) {
        out16[idx] = (_Float16)gelu_f(v);
      } else {
        float bv = bias ? bias[col] : 0.0f;
        outF[idx] += v + bv;
      }
    }
  }
}

// ---------------- per-(b,c) sum of squares over n (for L2 norms) -----------
__global__ void k_sumsq(const _Float16* __restrict__ q,
                        const _Float16* __restrict__ k, float* __restrict__ nq,
                        float* __restrict__ nk) {
  int b = blockIdx.y;
  int n0 = blockIdx.x * 64;
  int c = threadIdx.x;
  float aq = 0.f, ak = 0.f;
  for (int r = 0; r < 64; r++) {
    size_t idx = ((size_t)b * N_ + n0 + r) * C_ + c;
    float qv = (float)q[idx];
    float kv = (float)k[idx];
    aq += qv * qv;
    ak += kv * kv;
  }
  atomicAdd(&nq[b * C_ + c], aq);
  atomicAdd(&nk[b * C_ + c], ak);
}

// ---------------- channel attention: attn = softmax(kTn . qTn^T * rescale) -
__global__ void k_attn(const _Float16* __restrict__ qh,
                       const _Float16* __restrict__ kh,
                       const float* __restrict__ nq, const float* __restrict__ nk,
                       const float* __restrict__ resc,
                       float* __restrict__ attnOut) {
  __shared__ __align__(16) _Float16 sk[64 * 32];
  __shared__ __align__(16) _Float16 sq[64 * 32];
  __shared__ float sat[32 * 32];
  int b = blockIdx.x >> 3, h = blockIdx.x & 7;
  int tid = threadIdx.x;
  int lr = tid >> 2, lc = (tid & 3) * 8;      // cooperative 64x32 tile loads
  int d = tid >> 3, e0 = (tid & 7) * 4;       // 4 (d,e) pairs per thread
  float acc[4] = {0.f, 0.f, 0.f, 0.f};
  for (int n0 = 0; n0 < N_; n0 += 64) {
    size_t g = ((size_t)b * N_ + n0 + lr) * C_ + h * 32 + lc;
    *(uint4*)(sk + lr * 32 + lc) = *(const uint4*)(kh + g);
    *(uint4*)(sq + lr * 32 + lc) = *(const uint4*)(qh + g);
    __syncthreads();
    for (int r = 0; r < 64; r++) {
      float kv = (float)sk[r * 32 + d];
#pragma unroll
      for (int j = 0; j < 4; j++) acc[j] += kv * (float)sq[r * 32 + e0 + j];
    }
    __syncthreads();
  }
  float rs = resc[h];
  float dk = fmaxf(sqrtf(nk[b * C_ + h * 32 + d]), 1e-12f);
#pragma unroll
  for (int j = 0; j < 4; j++) {
    int e = e0 + j;
    float dq = fmaxf(sqrtf(nq[b * C_ + h * 32 + e]), 1e-12f);
    sat[d * 32 + e] = acc[j] * rs / (dk * dq);
  }
  __syncthreads();
  if (tid < 32) {
    float mx = -1e30f;
    for (int e = 0; e < 32; e++) mx = fmaxf(mx, sat[tid * 32 + e]);
    float ex[32], sum = 0.f;
    for (int e = 0; e < 32; e++) {
      ex[e] = expf(sat[tid * 32 + e] - mx);
      sum += ex[e];
    }
    float inv = 1.0f / sum;
    for (int e = 0; e < 32; e++)
      attnOut[(((size_t)b * HEADS_ + h) * 32 + tid) * 32 + e] = ex[e] * inv;
  }
}

// ---------------- apply attn per token: o[h,d] = sum_e attn[h,d,e]*v[h,e] --
__global__ void k_ovec(const _Float16* __restrict__ vh,
                       const float* __restrict__ attn,
                       _Float16* __restrict__ oh) {
  __shared__ float sat[HEADS_ * 32 * 32];
  int b = blockIdx.y;
  for (int i = threadIdx.x; i < HEADS_ * 32 * 32; i += 256)
    sat[i] = attn[(size_t)b * HEADS_ * 32 * 32 + i];
  __syncthreads();
  int t = blockIdx.x * 32 + (threadIdx.x >> 3);
  int h = threadIdx.x & 7;
  size_t base = ((size_t)b * N_ + t) * C_ + h * 32;
  float vv[32];
#pragma unroll
  for (int q = 0; q < 4; q++) {
    h8 x = *(const h8*)(vh + base + q * 8);
#pragma unroll
    for (int j = 0; j < 8; j++) vv[q * 8 + j] = (float)x[j];
  }
  const float* ah = sat + h * 1024;
  __align__(16) _Float16 ov[32];
  for (int dd = 0; dd < 32; dd++) {
    float s = 0.f;
#pragma unroll
    for (int e = 0; e < 32; e++) s += ah[dd * 32 + e] * vv[e];
    ov[dd] = (_Float16)s;
  }
#pragma unroll
  for (int q = 0; q < 4; q++) *(h8*)(oh + base + q * 8) = *(const h8*)(ov + q * 8);
}

// ---------------- LayerNorm over channels, write f16 -----------------------
__global__ void k_ln(const float* __restrict__ X, const float* __restrict__ g,
                     const float* __restrict__ bb, _Float16* __restrict__ Y) {
  int lane = threadIdx.x & 31, wave = threadIdx.x >> 5;
  size_t token = (size_t)blockIdx.x * 8 + wave;
  const float* xp = X + token * C_ + lane * 8;
  float4 u0 = *(const float4*)xp;
  float4 u1 = *(const float4*)(xp + 4);
  float v[8] = {u0.x, u0.y, u0.z, u0.w, u1.x, u1.y, u1.z, u1.w};
  float s = 0.f;
#pragma unroll
  for (int i = 0; i < 8; i++) s += v[i];
#pragma unroll
  for (int off = 16; off > 0; off >>= 1) s += __shfl_xor(s, off, 32);
  float mu = s * (1.0f / 256.0f);
  float sq = 0.f;
#pragma unroll
  for (int i = 0; i < 8; i++) {
    float d = v[i] - mu;
    sq += d * d;
  }
#pragma unroll
  for (int off = 16; off > 0; off >>= 1) sq += __shfl_xor(sq, off, 32);
  float inv = rsqrtf(sq * (1.0f / 256.0f) + 1e-5f);
  int c0 = lane * 8;
  h8 o;
#pragma unroll
  for (int i = 0; i < 8; i++)
    o[i] = (_Float16)((v[i] - mu) * inv * g[c0 + i] + bb[c0 + i]);
  *(h8*)(Y + token * C_ + c0) = o;
}

// ---------------- depthwise 3x3, channels-last, SAME pad -------------------
// MODE 0: GELU -> f16 out   MODE 1: accumulate into f32 (residual)
template <int MODE>
__global__ void k_dw3x3(const _Float16* __restrict__ in,
                        const float* __restrict__ w9,
                        _Float16* __restrict__ out16, float* __restrict__ outF,
                        int Cc) {
  int p = blockIdx.x;
  int y = p >> 7, x = p & (W_ - 1);
  int b = blockIdx.y;
  for (int c = threadIdx.x; c < Cc; c += 256) {
    float acc = 0.f;
#pragma unroll
    for (int dy = -1; dy <= 1; dy++) {
      int yy = y + dy;
      if (yy < 0 || yy >= H_) continue;
#pragma unroll
      for (int dx = -1; dx <= 1; dx++) {
        int xx = x + dx;
        if (xx < 0 || xx >= W_) continue;
        acc += (float)in[(((size_t)b * H_ + yy) * W_ + xx) * Cc + c] *
               w9[c * 9 + (dy + 1) * 3 + (dx + 1)];
      }
    }
    size_t oi = ((size_t)b * N_ + p) * Cc + c;
    if (MODE == 0)
      out16[oi] = (_Float16)gelu_f(acc);
    else
      outF[oi] += acc;
  }
}

// ===========================================================================
extern "C" void kernel_launch(void* const* d_in, const int* in_sizes, int n_in,
                              void* d_out, int out_size, void* d_ws,
                              size_t ws_size, hipStream_t stream) {
  (void)in_sizes; (void)n_in; (void)out_size; (void)ws_size;
  const float* x      = (const float*)d_in[0];
  const float* illu   = (const float*)d_in[1];
  const float* Wq     = (const float*)d_in[2];
  const float* Wk     = (const float*)d_in[3];
  const float* Wv     = (const float*)d_in[4];
  const float* resc   = (const float*)d_in[5];
  const float* Wproj  = (const float*)d_in[6];
  const float* bproj  = (const float*)d_in[7];
  const float* pos1   = (const float*)d_in[8];
  const float* pos2   = (const float*)d_in[9];
  const float* ln_g   = (const float*)d_in[10];
  const float* ln_b   = (const float*)d_in[11];
  const float* ff1    = (const float*)d_in[12];
  const float* ffdw   = (const float*)d_in[13];
  const float* ff3    = (const float*)d_in[14];
  float* out = (float*)d_out;

  char* base = (char*)d_ws;
  size_t off = 0;
  auto alloc = [&](size_t bytes) -> char* {
    off = (off + 255) & ~(size_t)255;
    char* p = base + off;
    off += bytes;
    return p;
  };

  float*     X      = (float*)alloc((size_t)BN_ * C_ * 4);   // bhwc f32 state
  _Float16*  xh     = (_Float16*)alloc((size_t)BN_ * C_ * 2); // also reused as oh
  _Float16*  qh     = (_Float16*)alloc((size_t)BN_ * C_ * 2); // reused as pos-mid
  _Float16*  kh     = (_Float16*)alloc((size_t)BN_ * C_ * 2); // reused as LN out
  _Float16*  vh     = (_Float16*)alloc((size_t)BN_ * C_ * 2);
  _Float16*  illu_h = (_Float16*)alloc((size_t)BN_ * C_ * 2);
  _Float16*  t1h    = (_Float16*)alloc((size_t)BN_ * MC_ * 2);
  _Float16*  t2h    = (_Float16*)alloc((size_t)BN_ * MC_ * 2);
  _Float16*  wqT    = (_Float16*)alloc((size_t)L_ * C_ * C_ * 2);
  _Float16*  wkT    = (_Float16*)alloc((size_t)L_ * C_ * C_ * 2);
  _Float16*  wvT    = (_Float16*)alloc((size_t)L_ * C_ * C_ * 2);
  _Float16*  wpT    = (_Float16*)alloc((size_t)L_ * C_ * C_ * 2);
  _Float16*  ff1h   = (_Float16*)alloc((size_t)L_ * MC_ * C_ * 2);
  _Float16*  ff3h   = (_Float16*)alloc((size_t)L_ * C_ * MC_ * 2);
  float*     normbuf= (float*)alloc((size_t)2 * B_ * C_ * 4);
  float*     nq     = normbuf;
  float*     nk     = normbuf + B_ * C_;
  float*     attnb  = (float*)alloc((size_t)B_ * HEADS_ * 32 * 32 * 4);

  dim3 tb(32, 8);

  // Input transposes: bchw -> bhwc (f32 state, f16 illumination)
  k_tr2d<float, float><<<dim3(N_ / 32, C_ / 32, B_), tb, 0, stream>>>(x, X, C_, N_);
  k_tr2d<float, _Float16><<<dim3(N_ / 32, C_ / 32, B_), tb, 0, stream>>>(illu, illu_h, C_, N_);

  // Weight prep: [in][out] -> [out][in] f16 (WMMA B operand layout)
  for (int l = 0; l < L_; l++) {
    size_t w = (size_t)l * C_ * C_;
    k_tr2d<float, _Float16><<<dim3(8, 8, 1), tb, 0, stream>>>(Wq + w, wqT + w, C_, C_);
    k_tr2d<float, _Float16><<<dim3(8, 8, 1), tb, 0, stream>>>(Wk + w, wkT + w, C_, C_);
    k_tr2d<float, _Float16><<<dim3(8, 8, 1), tb, 0, stream>>>(Wv + w, wvT + w, C_, C_);
    k_tr2d<float, _Float16><<<dim3(8, 8, 1), tb, 0, stream>>>(Wproj + w, wpT + w, C_, C_);
  }
  // ff1/ff3 are already [out][in] (OIHW 1x1): straight cast
  k_cast<<<(L_ * MC_ * C_) / 256, 256, 0, stream>>>(ff1, ff1h, L_ * MC_ * C_);
  k_cast<<<(L_ * C_ * MC_) / 256, 256, 0, stream>>>(ff3, ff3h, L_ * C_ * MC_);

  dim3 gC(C_ / 64, BN_ / 128);    // GEMM grids: 16x64 per wave, 8 waves/block
  dim3 gMC(MC_ / 64, BN_ / 128);

  for (int l = 0; l < L_; l++) {
    const _Float16* wq_l  = wqT + (size_t)l * C_ * C_;
    const _Float16* wk_l  = wkT + (size_t)l * C_ * C_;
    const _Float16* wv_l  = wvT + (size_t)l * C_ * C_;
    const _Float16* wp_l  = wpT + (size_t)l * C_ * C_;
    const _Float16* ff1_l = ff1h + (size_t)l * MC_ * C_;
    const _Float16* ff3_l = ff3h + (size_t)l * C_ * MC_;
    const float* pos1_l = pos1 + (size_t)l * C_ * 9;
    const float* pos2_l = pos2 + (size_t)l * C_ * 9;
    const float* dw_l   = ffdw + (size_t)l * MC_ * 9;
    const float* bias_l = bproj + (size_t)l * C_;
    const float* g_l    = ln_g + (size_t)l * C_;
    const float* b_l    = ln_b + (size_t)l * C_;
    const float* resc_l = resc + (size_t)l * HEADS_;

    // A operand in f16
    k_cast<<<(BN_ * C_) / 256, 256, 0, stream>>>(X, xh, BN_ * C_);

    // q, k, v = x @ W ; v *= illu (fused)
    k_gemm_wmma<0><<<gC, 256, 0, stream>>>(xh, wq_l, nullptr, nullptr, qh, nullptr, C_, C_);
    k_gemm_wmma<0><<<gC, 256, 0, stream>>>(xh, wk_l, nullptr, nullptr, kh, nullptr, C_, C_);
    k_gemm_wmma<1><<<gC, 256, 0, stream>>>(xh, wv_l, illu_h, nullptr, vh, nullptr, C_, C_);

    // L2 norms over n, channel attention, apply attention
    hipMemsetAsync(normbuf, 0, (size_t)2 * B_ * C_ * 4, stream);
    k_sumsq<<<dim3(N_ / 64, B_), 256, 0, stream>>>(qh, kh, nq, nk);
    k_attn<<<B_ * HEADS_, 256, 0, stream>>>(qh, kh, nq, nk, resc_l, attnb);
    k_ovec<<<dim3(N_ / 32, B_), 256, 0, stream>>>(vh, attnb, xh /* oh */);

    // X += o @ Wproj + bproj (residual fused into epilogue)
    k_gemm_wmma<3><<<gC, 256, 0, stream>>>(xh, wp_l, nullptr, bias_l, nullptr, X, C_, C_);

    // positional path: X += dw(gelu(dw(v)))
    k_dw3x3<0><<<dim3(N_, B_), 256, 0, stream>>>(vh, pos1_l, qh, nullptr, C_);
    k_dw3x3<1><<<dim3(N_, B_), 256, 0, stream>>>(qh, pos2_l, nullptr, X, C_);

    // FF: LN -> 1x1 (GELU) -> dw3x3 (GELU) -> 1x1, X += result
    k_ln<<<BN_ / 8, 256, 0, stream>>>(X, g_l, b_l, kh /* yh */);
    k_gemm_wmma<2><<<gMC, 256, 0, stream>>>(kh, ff1_l, nullptr, nullptr, t1h, nullptr, C_, MC_);
    k_dw3x3<0><<<dim3(N_, B_), 256, 0, stream>>>(t1h, dw_l, t2h, nullptr, MC_);
    k_gemm_wmma<3><<<gC, 256, 0, stream>>>(t2h, ff3_l, nullptr, nullptr, nullptr, X, MC_, C_);
  }

  // Output transpose: bhwc -> bchw
  k_tr2d<float, float><<<dim3(C_ / 32, N_ / 32, B_), tb, 0, stream>>>(X, out, N_, C_);
}
